// MultiheadSelfAttention_11218454577207
// MI455X (gfx1250) — compile-verified, matrix operand order checked by reading, not run on hardware
//
#include <hip/hip_runtime.h>
#include <hip/hip_bf16.h>

typedef __attribute__((ext_vector_type(16))) _Float16 v16h;
typedef __attribute__((ext_vector_type(8)))  _Float16 v8h;
typedef __attribute__((ext_vector_type(8)))  float    v8f;

#define N_NODES 4096
#define DMODEL  512
#define NHEAD   8
#define HDIM    64

__device__ __forceinline__ v16h make_v16(v8h lo, v8h hi) {
  v16h r;
#pragma unroll
  for (int i = 0; i < 8; ++i) { r[i] = lo[i]; r[i + 8] = hi[i]; }
  return r;
}

// Load one WMMA f16 operand tile (16 rows x 32 K) from a row-major f16 matrix
// (global or LDS). A operand: rows = M rows. B operand: pass B^T (rows = N cols).
// ISA layout: lanes 0-15 -> K {0..7, 16..23}; lanes 16-31 -> K {8..15, 24..31}.
__device__ __forceinline__ v16h load_tile(const _Float16* p, int row0, int ld,
                                          int k0, int lane) {
  int r    = row0 + (lane & 15);
  int koff = (lane >> 4) << 3;
  const _Float16* rp = p + (size_t)r * ld + k0 + koff;
  v8h lo = *(const v8h*)(rp);
  v8h hi = *(const v8h*)(rp + 16);
  return make_v16(lo, hi);
}

__device__ __forceinline__ v8f wmma16(v16h a, v16h b, v8f c) {
  return __builtin_amdgcn_wmma_f32_16x16x32_f16(false, a, false, b,
                                                (short)0, c, false, false);
}

__device__ __forceinline__ v8f vzero8() {
  v8f z;
#pragma unroll
  for (int i = 0; i < 8; ++i) z[i] = 0.0f;
  return z;
}

// ---------------- stage 0: conversions ----------------

__global__ __launch_bounds__(256) void cvt_x_kernel(const float* __restrict__ x,
                                                    _Float16* __restrict__ xh) {
  int i = blockIdx.x * 256 + threadIdx.x;  // 4096*512 total
  xh[i] = (_Float16)x[i];
}

// Wt[cout][cin] with cout = h*64 + d  taken from original column d*8 + h.
__global__ __launch_bounds__(256) void cvt_w_kernel(const float* __restrict__ w,
                                                    _Float16* __restrict__ wt) {
  int i = blockIdx.x * 256 + threadIdx.x;  // 512*512 total
  int cout = i >> 9;
  int c    = i & 511;
  int h    = cout >> 6;
  int d    = cout & 63;
  wt[i] = (_Float16)w[c * DMODEL + d * NHEAD + h];
}

// ---------------- stage 1: projections ----------------
// Out[n][cout] = sum_c Xh[n][c] * Wt[cout][c].  One wave -> 16x64 tile.
// All operand tiles for a K-step are fetched before the 4 WMMAs so the
// scheduler can stagger s_wait_loadcnt instead of draining per tile.
__global__ __launch_bounds__(256) void proj_kernel(const _Float16* __restrict__ xh,
                                                   const _Float16* __restrict__ wt,
                                                   _Float16* __restrict__ out,
                                                   int transposed, float scale) {
  int lane = threadIdx.x & 31;
  int wv   = threadIdx.x >> 5;
  int task = blockIdx.x * 8 + wv;      // 0..2047
  int rt   = task >> 3;                // 256 row tiles
  int cg   = task & 7;                 // 8 col groups of 64
  int row0  = rt * 16;
  int cout0 = cg * 64;

  v8f acc[4];
#pragma unroll
  for (int t = 0; t < 4; ++t) acc[t] = vzero8();

  for (int c0 = 0; c0 < DMODEL; c0 += 32) {
    v16h a = load_tile(xh, row0, DMODEL, c0, lane);
    v16h b[4];
#pragma unroll
    for (int t = 0; t < 4; ++t)
      b[t] = load_tile(wt, cout0 + t * 16, DMODEL, c0, lane);
#pragma unroll
    for (int t = 0; t < 4; ++t)
      acc[t] = wmma16(a, b[t], acc[t]);
  }

  int hb = lane >> 4;
  int nl = lane & 15;
  if (!transposed) {
#pragma unroll
    for (int t = 0; t < 4; ++t) {
#pragma unroll
      for (int i = 0; i < 8; ++i) {
        int r  = row0 + i + 8 * hb;
        int cc = cout0 + t * 16 + nl;
        out[(size_t)r * DMODEL + cc] = (_Float16)(acc[t][i] * scale);
      }
    }
  } else {
#pragma unroll
    for (int t = 0; t < 4; ++t) {
      v8h pk;
#pragma unroll
      for (int i = 0; i < 8; ++i) pk[i] = (_Float16)(acc[t][i] * scale);
      int cc = cout0 + t * 16 + nl;
      *(v8h*)(out + (size_t)cc * N_NODES + row0 + 8 * hb) = pk;  // contiguous n
    }
  }
}

// ---------------- stage 2: flash attention ----------------
// Block = 1 head x 128 query rows (8 waves x 16 rows). K/V^T tiles for the
// current 32 keys are staged in LDS once per block (8x traffic reduction),
// double-buffered: global->reg loads issue at the top of the iteration,
// ds-stores land in the spare buffer after compute, one barrier per iteration.
__global__ __launch_bounds__(256) void attn_kernel(const _Float16* __restrict__ Qh,
                                                   const _Float16* __restrict__ Kh,
                                                   const _Float16* __restrict__ Vt,
                                                   float* __restrict__ out) {
  __shared__ _Float16 KT[2][32][72];  // keys x d   (rows padded: 72 halves)
  __shared__ _Float16 VT[2][64][40];  // d    x keys (rows padded: 40 halves)
  __shared__ _Float16 PB[8][16][40];  // per-wave P staging

  int tid  = threadIdx.x;
  int lane = tid & 31;
  int wv   = tid >> 5;
  int h    = blockIdx.x >> 5;          // 8 heads
  int g    = blockIdx.x & 31;          // 32 query groups of 128
  int q0   = g * 128 + wv * 16;

  const _Float16* Qhead = Qh + h * HDIM;                      // [4096][512] slice
  const _Float16* Khead = Kh + h * HDIM;
  const _Float16* Vhead = Vt + (size_t)(h * HDIM) * N_NODES;  // [64][4096]

  // per-thread staging assignment: one b128 of K tile + one b128 of V tile
  int krow = tid >> 3, kseg = tid & 7;  // K tile: 32 rows x 128B
  int vrow = tid >> 2, vseg = tid & 3;  // V tile: 64 rows x  64B

  // preload tile 0
  {
    v8h kr = *(const v8h*)(Khead + (size_t)krow * DMODEL + kseg * 8);
    v8h vr = *(const v8h*)(Vhead + (size_t)vrow * N_NODES + vseg * 8);
    *(v8h*)(&KT[0][krow][kseg * 8]) = kr;
    *(v8h*)(&VT[0][vrow][vseg * 8]) = vr;
  }

  v16h qa0 = load_tile(Qhead, q0, DMODEL, 0, lane);   // d = 0..31
  v16h qa1 = load_tile(Qhead, q0, DMODEL, 32, lane);  // d = 32..63

  v8f o[4];
#pragma unroll
  for (int t = 0; t < 4; ++t) o[t] = vzero8();
  float mrow[8], lrow[8];
#pragma unroll
  for (int i = 0; i < 8; ++i) { mrow[i] = -__builtin_inff(); lrow[i] = 0.0f; }

  int hb = lane >> 4;
  int nl = lane & 15;
  _Float16* pw = &PB[wv][0][0];

  __syncthreads();

  for (int it = 0; it < N_NODES / 32; ++it) {
    int cur = it & 1;
    int kb  = it * 32;
    bool more = (it + 1) < (N_NODES / 32);

    // issue next tile's global loads early (consumed after compute)
    v8h kr, vr;
    if (more) {
      kr = *(const v8h*)(Khead + (size_t)(kb + 32 + krow) * DMODEL + kseg * 8);
      vr = *(const v8h*)(Vhead + (size_t)vrow * N_NODES + (kb + 32) + vseg * 8);
    }

    const _Float16* kt = &KT[cur][0][0];
    const _Float16* vt = &VT[cur][0][0];

    // S = Q K^T  (Q pre-scaled by 1/8)
    v8f s0 = vzero8(), s1 = vzero8();
    {
      v16h k0a = load_tile(kt, 0,  72, 0,  lane);
      v16h k0b = load_tile(kt, 0,  72, 32, lane);
      v16h k1a = load_tile(kt, 16, 72, 0,  lane);
      v16h k1b = load_tile(kt, 16, 72, 32, lane);
      s0 = wmma16(qa0, k0a, s0);
      s0 = wmma16(qa1, k0b, s0);
      s1 = wmma16(qa0, k1a, s1);
      s1 = wmma16(qa1, k1b, s1);
    }

    // diagonal mask (self-loop = -inf)
    if ((q0 >> 5) == it) {
#pragma unroll
      for (int i = 0; i < 8; ++i) {
        int m = q0 + i + 8 * hb;
        if (m == kb + nl)      s0[i] = -__builtin_inff();
        if (m == kb + 16 + nl) s1[i] = -__builtin_inff();
      }
    }

    // online softmax: rows live in slot i across 16 lanes of each half-wave
#pragma unroll
    for (int i = 0; i < 8; ++i) {
      float v = fmaxf(s0[i], s1[i]);
#pragma unroll
      for (int off = 1; off < 16; off <<= 1) v = fmaxf(v, __shfl_xor(v, off, 32));
      float mn = fmaxf(mrow[i], v);
      float sc = __expf(mrow[i] - mn);
      float p0 = __expf(s0[i] - mn);
      float p1 = __expf(s1[i] - mn);
      float rs = p0 + p1;
#pragma unroll
      for (int off = 1; off < 16; off <<= 1) rs += __shfl_xor(rs, off, 32);
      lrow[i] = lrow[i] * sc + rs;
      mrow[i] = mn;
#pragma unroll
      for (int t = 0; t < 4; ++t) o[t][i] *= sc;

      int m = i + 8 * hb;
      pw[m * 40 + nl]      = (_Float16)p0;
      pw[m * 40 + nl + 16] = (_Float16)p1;
    }

    // reload P in A-operand layout (wave-private LDS; DS ops in-order per wave)
    {
      int am   = lane & 15;
      int koff = (lane >> 4) << 3;
      v8h lo = *(const v8h*)(pw + am * 40 + koff);
      v8h hi = *(const v8h*)(pw + am * 40 + 16 + koff);
      v16h pa = make_v16(lo, hi);
#pragma unroll
      for (int t = 0; t < 4; ++t) {
        v16h vb = load_tile(vt, t * 16, 40, 0, lane);  // V^T rows = d cols
        o[t] = wmma16(pa, vb, o[t]);
      }
    }

    // publish next tile into the spare buffer
    if (more) {
      *(v8h*)(&KT[cur ^ 1][krow][kseg * 8]) = kr;
      *(v8h*)(&VT[cur ^ 1][vrow][vseg * 8]) = vr;
    }
    __syncthreads();
  }

  // epilogue: normalize, tanh, scatter to interleaved layout col = d*8 + h
#pragma unroll
  for (int i = 0; i < 8; ++i) {
    float inv = 1.0f / lrow[i];
    int m = q0 + i + 8 * hb;
#pragma unroll
    for (int t = 0; t < 4; ++t) {
      int d = t * 16 + nl;
      out[(size_t)m * DMODEL + d * NHEAD + h] = tanhf(o[t][i] * inv);
    }
  }
}

// ---------------- launcher ----------------

extern "C" void kernel_launch(void* const* d_in, const int* in_sizes, int n_in,
                              void* d_out, int out_size, void* d_ws, size_t ws_size,
                              hipStream_t stream) {
  (void)in_sizes; (void)n_in; (void)out_size; (void)ws_size;
  const float* x  = (const float*)d_in[0];
  const float* Wq = (const float*)d_in[1];
  const float* Wk = (const float*)d_in[2];
  const float* Wv = (const float*)d_in[3];
  float* out = (float*)d_out;

  _Float16* ws  = (_Float16*)d_ws;
  _Float16* Xh  = ws;                      // 4096*512
  _Float16* Wtq = Xh  + N_NODES * DMODEL;  // 512*512
  _Float16* Wtk = Wtq + DMODEL * DMODEL;
  _Float16* Wtv = Wtk + DMODEL * DMODEL;
  _Float16* Qb  = Wtv + DMODEL * DMODEL;   // 4096*512 (head-permuted, pre-scaled)
  _Float16* Kb  = Qb  + N_NODES * DMODEL;  // 4096*512
  _Float16* Vtb = Kb  + N_NODES * DMODEL;  // 512*4096 (V^T)

  cvt_x_kernel<<<(N_NODES * DMODEL) / 256, 256, 0, stream>>>(x, Xh);
  cvt_w_kernel<<<(DMODEL * DMODEL) / 256, 256, 0, stream>>>(Wq, Wtq);
  cvt_w_kernel<<<(DMODEL * DMODEL) / 256, 256, 0, stream>>>(Wk, Wtk);
  cvt_w_kernel<<<(DMODEL * DMODEL) / 256, 256, 0, stream>>>(Wv, Wtv);

  // 2048 wave-tasks per projection -> 256 blocks of 8 waves
  proj_kernel<<<256, 256, 0, stream>>>(Xh, Wtq, Qb,  0, 0.125f);  // Q * 1/sqrt(64)
  proj_kernel<<<256, 256, 0, stream>>>(Xh, Wtk, Kb,  0, 1.0f);
  proj_kernel<<<256, 256, 0, stream>>>(Xh, Wtv, Vtb, 1, 1.0f);

  // block = (head, 128 query rows): 8 heads * 32 groups = 256 blocks
  attn_kernel<<<256, 256, 0, stream>>>(Qb, Kb, Vtb, out);
}
